// LinkPredictionModel_730144441189
// MI455X (gfx1250) — matrix operand order repaired and checked by reading, not compiled, not run on hardware
//
#include <hip/hip_runtime.h>
#include <hip/hip_bf16.h>

#define D 64            // feature width (D_IN == D_HID == 64)
#define DV 16           // D/4 float4 chunks per row

typedef __attribute__((ext_vector_type(16))) __bf16 v16bf;
typedef __attribute__((ext_vector_type(8)))  float  v8f;

static __device__ __forceinline__ __bf16 f2bf(float x) { return (__bf16)x; }

// ---------------- degree / normalization ----------------
__global__ void k_deg_init(float* deg, int N) {
    int i = blockIdx.x * blockDim.x + threadIdx.x;
    if (i < N) deg[i] = 1.0f;                       // self-loop contributes 1
}

__global__ void k_deg_count(const long long* __restrict__ ei, float* deg, int E) {
    int e = blockIdx.x * blockDim.x + threadIdx.x;
    if (e < E) {
        long long d = ei[(long long)E + e];         // dst row
        unsafeAtomicAdd(&deg[(int)d], 1.0f);
    }
}

__global__ void k_deg_to_dinv(float* deg, int N) {
    int i = blockIdx.x * blockDim.x + threadIdx.x;
    if (i < N) deg[i] = rsqrtf(deg[i]);             // deg >= 1 always
}

// ---------------- WMMA GEMM: T[N x 64] = X[N x 64] @ W[64 x 64] ----------------
// One wave computes a 16x16 tile. blockDim = 128 (4 waves = 4 column tiles).
// A layout (16-bit 16x32, ISA 7.12.2): lane row = lane%16;
//   lanes 0-15 hold K = {0..7, 16..23}, lanes 16-31 hold K = {8..15, 24..31}.
// B layout (32x16): col = lane%16; lanes 0-15 hold K={0..7,16..23}, lanes 16-31 K={8..15,24..31} (same as A).
// D layout (f32 16x16): VGPR v -> m = v + 8*(lane/16), n = lane%16.
__global__ __launch_bounds__(128) void k_gemm_wmma(
    const float* __restrict__ X, const float* __restrict__ W,
    float* __restrict__ T, int N)
{
    const int lane = threadIdx.x & 31;
    const int wid  = threadIdx.x >> 5;      // column tile 0..3
    const int m0   = blockIdx.x * 16;
    const int n0   = wid * 16;
    const int half = lane >> 4;
    const int idx  = lane & 15;

    const float* arow = X + (size_t)(m0 + idx) * D;
    v8f c = {};

#pragma unroll
    for (int kc = 0; kc < 2; ++kc) {
        const int kk = kc * 32;
        v16bf a, b;

        // A: two contiguous 8-float runs for this lane's row
        const float4* p0 = (const float4*)(arow + kk + 8 * half);
        const float4* p1 = (const float4*)(arow + kk + 16 + 8 * half);
        float4 r0 = p0[0], r1 = p0[1], r2 = p1[0], r3 = p1[1];
        a[0]  = f2bf(r0.x); a[1]  = f2bf(r0.y); a[2]  = f2bf(r0.z); a[3]  = f2bf(r0.w);
        a[4]  = f2bf(r1.x); a[5]  = f2bf(r1.y); a[6]  = f2bf(r1.z); a[7]  = f2bf(r1.w);
        a[8]  = f2bf(r2.x); a[9]  = f2bf(r2.y); a[10] = f2bf(r2.z); a[11] = f2bf(r2.w);
        a[12] = f2bf(r3.x); a[13] = f2bf(r3.y); a[14] = f2bf(r3.z); a[15] = f2bf(r3.w);

        // B: column n0+idx of W (row-major 64x64), K = kk + 8*half + (e&7) + 16*(e>>3)
#pragma unroll
        for (int e = 0; e < 16; ++e) {
            int k = kk + 8 * half + (e & 7) + ((e >> 3) << 4);
            b[e] = f2bf(W[(size_t)k * D + n0 + idx]);
        }

        c = __builtin_amdgcn_wmma_f32_16x16x32_bf16(
                /*neg_a=*/false, a, /*neg_b=*/false, b,
                /*c_mod=*/(short)0, c, /*reuse_a=*/false, /*reuse_b=*/false);
    }

#pragma unroll
    for (int v = 0; v < 8; ++v) {
        int m = m0 + v + 8 * half;
        T[(size_t)m * D + n0 + idx] = c[v];
    }
}

// ---------------- ACC = T * dinv^2 (self-loop) + bias ----------------
__global__ void k_init_acc(const float* __restrict__ T, const float* __restrict__ dinv,
                           const float* __restrict__ bias, float* __restrict__ acc, int N)
{
    int t = blockIdx.x * blockDim.x + threadIdx.x;   // over N * DV
    if (t >= N * DV) return;
    int node = t >> 4, q = t & 15;
    float s = dinv[node]; s *= s;
    float4 v = ((const float4*)T)[(size_t)node * DV + q];
    float4 b = ((const float4*)bias)[q];
    float4 r = { v.x * s + b.x, v.y * s + b.y, v.z * s + b.z, v.w * s + b.w };
    ((float4*)acc)[(size_t)node * DV + q] = r;
}

// ---------------- edge scatter: ACC[dst] += T[src] * dinv[src]*dinv[dst] ----------------
__global__ void k_scatter(const long long* __restrict__ ei,
                          const float* __restrict__ T, const float* __restrict__ dinv,
                          float* __restrict__ acc, int E)
{
    int t = blockIdx.x * blockDim.x + threadIdx.x;   // over E * DV
    if (t >= E * DV) return;
    int e = t >> 4, q = t & 15;
    int s = (int)ei[e];
    int d = (int)ei[(long long)E + e];
    float w = dinv[s] * dinv[d];
    float4 v = ((const float4*)T)[(size_t)s * DV + q];
    float* out = acc + (size_t)d * D + q * 4;
    unsafeAtomicAdd(out + 0, v.x * w);
    unsafeAtomicAdd(out + 1, v.y * w);
    unsafeAtomicAdd(out + 2, v.z * w);
    unsafeAtomicAdd(out + 3, v.w * w);
}

// ---------------- ReLU: dst = max(acc, 0) ----------------
__global__ void k_relu(const float* __restrict__ acc, float* __restrict__ dst, int n4)
{
    int t = blockIdx.x * blockDim.x + threadIdx.x;   // over N * DV
    if (t >= n4) return;
    float4 v = ((const float4*)acc)[t];
    float4 r = { fmaxf(v.x, 0.f), fmaxf(v.y, 0.f), fmaxf(v.z, 0.f), fmaxf(v.w, 0.f) };
    ((float4*)dst)[t] = r;
}

static inline int cdiv(int a, int b) { return (a + b - 1) / b; }

extern "C" void kernel_launch(void* const* d_in, const int* in_sizes, int n_in,
                              void* d_out, int out_size, void* d_ws, size_t ws_size,
                              hipStream_t stream)
{
    const float* x  = (const float*)d_in[0];
    const float* W1 = (const float*)d_in[1];
    const float* b1 = (const float*)d_in[2];
    const float* W2 = (const float*)d_in[3];
    const float* b2 = (const float*)d_in[4];
    const float* W3 = (const float*)d_in[5];
    const float* b3 = (const float*)d_in[6];
    const long long* ei = (const long long*)d_in[7];

    const int N = in_sizes[0] / D;       // 100000
    const int E = in_sizes[7] / 2;       // 1000000

    // workspace partition: dinv[N] | T[N*D] | H[N*D]
    char* ws = (char*)d_ws;
    size_t off = 0;
    float* dinv = (float*)(ws + off); off += ((size_t)N * 4 + 255) & ~(size_t)255;
    float* T    = (float*)(ws + off); off += ((size_t)N * D * 4 + 255) & ~(size_t)255;
    float* H    = (float*)(ws + off);
    float* acc  = (float*)d_out;        // per-layer accumulator; final layer leaves result

    const int TB = 256;

    // degrees -> dinv
    k_deg_init   <<<cdiv(N, TB), TB, 0, stream>>>(dinv, N);
    k_deg_count  <<<cdiv(E, TB), TB, 0, stream>>>(ei, dinv, E);
    k_deg_to_dinv<<<cdiv(N, TB), TB, 0, stream>>>(dinv, N);

    const float* Ws[3] = { W1, W2, W3 };
    const float* bs[3] = { b1, b2, b3 };

    const float* Hin = x;
    for (int layer = 0; layer < 3; ++layer) {
        // T = Hin @ W  (WMMA bf16 -> f32)
        k_gemm_wmma<<<N / 16, 128, 0, stream>>>(Hin, Ws[layer], T, N);
        // ACC = T * dinv^2 + b
        k_init_acc <<<cdiv(N * DV, TB), TB, 0, stream>>>(T, dinv, bs[layer], acc, N);
        // ACC[dst] += T[src] * dinv[src]*dinv[dst]
        k_scatter  <<<cdiv(E * DV, TB), TB, 0, stream>>>(ei, T, dinv, acc, E);
        // ReLU
        float* dst = (layer < 2) ? H : (float*)d_out;   // last layer: in-place on d_out
        k_relu     <<<cdiv(N * DV, TB), TB, 0, stream>>>(acc, dst, N * DV);
        Hin = H;
    }
}
